// MultiHeadAttention_70781061038444
// MI455X (gfx1250) — compile-verified, matrix operand order checked by reading, not run on hardware
//
#include <hip/hip_runtime.h>
#include <hip/hip_bf16.h>

// ---------------------------------------------------------------------------
// MHA forward on gfx1250 (CDNA5, wave32) via v_wmma_f32_16x16x32_f16.
// Pipeline: f32->f16 pre-convert -> fused QKV GEMM (16x64/wave, ping-pong
// pipelined) -> flash attention -> output GEMM (16x64/wave, ping-pong).
// ---------------------------------------------------------------------------

typedef __attribute__((ext_vector_type(16))) _Float16 v16h;
typedef __attribute__((ext_vector_type(8)))  _Float16 v8h;
typedef __attribute__((ext_vector_type(8)))  float    v8f;
typedef __attribute__((ext_vector_type(4)))  float    v4f;

#define BSZ   2
#define QLEN  2048
#define DIM   1024
#define NH    16
#define DPH   64

// A/B fragment layout for V_WMMA_*_16X16X32_F16 (wave32, ISA 7.12.2):
//   lane L (0..15):  row/col = L, halves 0..7 = K {0..7},  8..15 = K {16..23}
//   lane L (16..31): row/col = L-16, halves 0..7 = K {8..15}, 8..15 = K {24..31}
// -> each lane reads two contiguous 8-half runs at +koff and +koff+16,
//    koff = (lane>>4)*8.
static __device__ inline v16h frag_from_f16(const _Float16* __restrict__ p) {
  v8h lo = *(const v8h*)(p);
  v8h hi = *(const v8h*)(p + 16);
  return __builtin_shufflevector(lo, hi, 0, 1, 2, 3, 4, 5, 6, 7,
                                 8, 9, 10, 11, 12, 13, 14, 15);
}

static __device__ inline v8f wmma_f16(v16h a, v16h b, v8f c) {
  // (neg_a, A, neg_b, B, c_mod, C, reuse_a, reuse_b)
  return __builtin_amdgcn_wmma_f32_16x16x32_f16(false, a, false, b,
                                                (short)0, c, false, false);
}

// ---------------------------------------------------------------------------
// Kernel 0: bulk f32 -> f16 conversion (bandwidth bound, runs once per array).
// ---------------------------------------------------------------------------
__global__ __launch_bounds__(256) void cvt_f16_kernel(
    const float* __restrict__ src, _Float16* __restrict__ dst, int n8) {
  const int i = blockIdx.x * 256 + threadIdx.x;
  if (i >= n8) return;
  const float* p = src + (size_t)i * 8;
  v4f a = *(const v4f*)(p);
  v4f b = *(const v4f*)(p + 4);
  v8h o;
#pragma unroll
  for (int j = 0; j < 4; ++j) {
    o[j]     = (_Float16)a[j];
    o[4 + j] = (_Float16)b[j];
  }
  *(v8h*)(dst + (size_t)i * 8) = o;
}

// ---------------------------------------------------------------------------
// Ping-pong pipelined 16x64 GEMM core:  acc[t] += A(16x1024) * B^T slice.
// arow: A row base (lane-resolved), wcol: B column base (lane-resolved).
// Set-0 fragments live across the backedge (loop phis -> no register moves);
// set-1 fragments are body-local.  3 independent WMMAs between accumulator
// reuse removes WMMA->WMMA hazard NOPs.
// ---------------------------------------------------------------------------
static __device__ inline void gemm_16x64(const _Float16* __restrict__ arow,
                                         const _Float16* __restrict__ wcol,
                                         v8f& acc0, v8f& acc1, v8f& acc2,
                                         v8f& acc3) {
  v16h a  = frag_from_f16(arow);
  v16h b0 = frag_from_f16(wcol);
  v16h b1 = frag_from_f16(wcol + 16 * DIM);
  v16h b2 = frag_from_f16(wcol + 32 * DIM);
  v16h b3 = frag_from_f16(wcol + 48 * DIM);
#pragma unroll 1
  for (int k = 0; k < DIM - 64; k += 64) {
    __builtin_prefetch(arow + k + 256, 0, 3);        // -> global_prefetch_b8
    // load set-1 (k+32)
    const v16h a2 = frag_from_f16(arow + k + 32);
    const v16h c0 = frag_from_f16(wcol + k + 32);
    const v16h c1 = frag_from_f16(wcol + 16 * DIM + k + 32);
    const v16h c2 = frag_from_f16(wcol + 32 * DIM + k + 32);
    const v16h c3 = frag_from_f16(wcol + 48 * DIM + k + 32);
    // compute set-0 (k)
    acc0 = wmma_f16(a, b0, acc0);
    acc1 = wmma_f16(a, b1, acc1);
    acc2 = wmma_f16(a, b2, acc2);
    acc3 = wmma_f16(a, b3, acc3);
    // reload set-0 (k+64) into the same loop-carried registers
    a  = frag_from_f16(arow + k + 64);
    b0 = frag_from_f16(wcol + k + 64);
    b1 = frag_from_f16(wcol + 16 * DIM + k + 64);
    b2 = frag_from_f16(wcol + 32 * DIM + k + 64);
    b3 = frag_from_f16(wcol + 48 * DIM + k + 64);
    // compute set-1 (k+32)
    acc0 = wmma_f16(a2, c0, acc0);
    acc1 = wmma_f16(a2, c1, acc1);
    acc2 = wmma_f16(a2, c2, acc2);
    acc3 = wmma_f16(a2, c3, acc3);
  }
  // tail: set-0 holds k = DIM-64; load and process k = DIM-32
  {
    const v16h a2 = frag_from_f16(arow + DIM - 32);
    const v16h c0 = frag_from_f16(wcol + DIM - 32);
    const v16h c1 = frag_from_f16(wcol + 16 * DIM + DIM - 32);
    const v16h c2 = frag_from_f16(wcol + 32 * DIM + DIM - 32);
    const v16h c3 = frag_from_f16(wcol + 48 * DIM + DIM - 32);
    acc0 = wmma_f16(a, b0, acc0);
    acc1 = wmma_f16(a, b1, acc1);
    acc2 = wmma_f16(a, b2, acc2);
    acc3 = wmma_f16(a, b3, acc3);
    acc0 = wmma_f16(a2, c0, acc0);
    acc1 = wmma_f16(a2, c1, acc1);
    acc2 = wmma_f16(a2, c2, acc2);
    acc3 = wmma_f16(a2, c3, acc3);
  }
}

// ---------------------------------------------------------------------------
// Kernel 1: fused Q/K/V projection, all-f16 operands, 16x64 tile per wave.
// Q pre-scaled by 1/8, stored [b][h][q][d]; K stored [b][h][k][d];
// V stored transposed [b][h][d][k] so the P@V GEMM gets contiguous B-frags.
// A wave's 64 columns == one head's full d-range (n0 is 64-aligned).
// ---------------------------------------------------------------------------
__global__ __launch_bounds__(256) void qkv_proj_kernel(
    const _Float16* __restrict__ Xh,
    const _Float16* __restrict__ Wqh, const float* __restrict__ bq,
    const _Float16* __restrict__ Wkh, const float* __restrict__ bk,
    const _Float16* __restrict__ Wvh, const float* __restrict__ bv,
    _Float16* __restrict__ qbuf, _Float16* __restrict__ kbuf,
    _Float16* __restrict__ vtbuf) {
  const int wave = threadIdx.x >> 5;
  const int lane = threadIdx.x & 31;
  int tile = blockIdx.x * 8 + wave;                 // 3 * 256 * 16 tiles
  const int mat = tile / (256 * 16);                // 0=Q 1=K 2=V
  tile %= (256 * 16);
  const int m0 = (tile >> 4) << 4;                  // row tile base (0..4080)
  const int n0 = (tile & 15) << 6;                  // col tile base, 64-wide

  const _Float16* W  = (mat == 0) ? Wqh : (mat == 1) ? Wkh : Wvh;
  const float* bias  = (mat == 0) ? bq  : (mat == 1) ? bk  : bv;

  const int r    = lane & 15;
  const int hs   = lane >> 4;
  const int koff = hs << 3;

  const _Float16* xrow = Xh + (size_t)(m0 + r) * DIM + koff;
  const _Float16* w0   = W  + (size_t)(n0 + r) * DIM + koff;

  v8f acc0 = {}, acc1 = {}, acc2 = {}, acc3 = {};
  gemm_16x64(xrow, w0, acc0, acc1, acc2, acc3);

  // Epilogue.  C layout: lane holds column (subtile*16 + r), rows m0+hs*8+j.
  const int h = n0 >> 6;                            // whole tile is one head
  auto store_tile = [&](v8f acc, int t) {
    const int n = n0 + t * 16 + r;
    const int d = (t << 4) + r;
    const float bn = bias[n];
#pragma unroll
    for (int j = 0; j < 8; ++j) {
      const int m    = m0 + (hs << 3) + j;          // 0..4095
      const int bidx = m >> 11;
      const int pos  = m & (QLEN - 1);
      const float v  = acc[j] + bn;
      const size_t bh = (size_t)(bidx * NH + h);
      if (mat == 0) {
        qbuf[(bh * QLEN + pos) * DPH + d] = (_Float16)(v * 0.125f);
      } else if (mat == 1) {
        kbuf[(bh * QLEN + pos) * DPH + d] = (_Float16)v;
      } else {
        vtbuf[(bh * DPH + d) * QLEN + pos] = (_Float16)v;
      }
    }
  };
  store_tile(acc0, 0);
  store_tile(acc1, 1);
  store_tile(acc2, 2);
  store_tile(acc3, 3);
}

// ---------------------------------------------------------------------------
// Kernel 2: flash attention, one wave per (b, h, 16-row q-tile).
// Per 32-wide k-step: S(16x32) = Q K^T (4 WMMAs), masked_fill(mask==0->1e-10),
// online softmax (xor-shuffle reductions within 16-lane groups), P staged
// through LDS (C-layout -> A-layout), then O(16x64) += P V (4 WMMAs).
// V fragments are loaded BEFORE the softmax chain so its VALU ops hide their
// latency.
// ---------------------------------------------------------------------------
__global__ __launch_bounds__(256) void attn_kernel(
    const _Float16* __restrict__ qbuf, const _Float16* __restrict__ kbuf,
    const _Float16* __restrict__ vtbuf, const int* __restrict__ mask,
    _Float16* __restrict__ ctx) {
  __shared__ __align__(16) _Float16 pbuf[8][16][32];   // 1 KB per wave

  const int wave = threadIdx.x >> 5;
  const int lane = threadIdx.x & 31;
  const int wid  = blockIdx.x * 8 + wave;              // 4096 waves
  const int qt = wid & 127;
  const int h  = (wid >> 7) & (NH - 1);
  const int b  = wid >> 11;
  const int q0 = qt << 4;

  const size_t bh = (size_t)(b * NH + h);
  const _Float16* Qbh  = qbuf  + bh * QLEN * DPH;
  const _Float16* Kbh  = kbuf  + bh * QLEN * DPH;
  const _Float16* VTbh = vtbuf + bh * DPH * QLEN;
  const int* mrow = mask + b * QLEN;

  const int r    = lane & 15;
  const int hs   = lane >> 4;
  const int koff = hs << 3;

  const v16h qa0 = frag_from_f16(Qbh + (size_t)(q0 + r) * DPH + koff);
  const v16h qa1 = frag_from_f16(Qbh + (size_t)(q0 + r) * DPH + 32 + koff);

  float mrun[8], lrun[8];
  v8f o0 = {}, o1 = {}, o2 = {}, o3 = {};
#pragma unroll
  for (int j = 0; j < 8; ++j) { mrun[j] = -3.0e38f; lrun[j] = 0.0f; }

#pragma unroll 1
  for (int kt = 0; kt < QLEN; kt += 32) {
    // ---- scores: two 16x16 subtiles ----
    v8f s0 = {}, s1 = {};
    {
      const _Float16* kp0 = Kbh + (size_t)(kt + r) * DPH + koff;
      s0 = wmma_f16(qa0, frag_from_f16(kp0), s0);
      s0 = wmma_f16(qa1, frag_from_f16(kp0 + 32), s0);
      const _Float16* kp1 = Kbh + (size_t)(kt + 16 + r) * DPH + koff;
      s1 = wmma_f16(qa0, frag_from_f16(kp1), s1);
      s1 = wmma_f16(qa1, frag_from_f16(kp1 + 32), s1);
    }
    // ---- V fragments: issue loads now, consumed after the softmax ----
    const v16h vb0 = frag_from_f16(VTbh + (size_t)( 0 + r) * QLEN + kt + koff);
    const v16h vb1 = frag_from_f16(VTbh + (size_t)(16 + r) * QLEN + kt + koff);
    const v16h vb2 = frag_from_f16(VTbh + (size_t)(32 + r) * QLEN + kt + koff);
    const v16h vb3 = frag_from_f16(VTbh + (size_t)(48 + r) * QLEN + kt + koff);

    // ---- masked_fill(mask==0 -> 1e-10), matching the reference ----
    const int mk0 = mrow[kt + r];
    const int mk1 = mrow[kt + 16 + r];
#pragma unroll
    for (int j = 0; j < 8; ++j) {
      if (mk0 == 0) s0[j] = 1e-10f;
      if (mk1 == 0) s1[j] = 1e-10f;
    }
    // ---- online softmax ----
    float p0[8], p1[8];
#pragma unroll
    for (int j = 0; j < 8; ++j) {
      float t = fmaxf(s0[j], s1[j]);
      t = fmaxf(t, __shfl_xor(t, 1, 32));
      t = fmaxf(t, __shfl_xor(t, 2, 32));
      t = fmaxf(t, __shfl_xor(t, 4, 32));
      t = fmaxf(t, __shfl_xor(t, 8, 32));
      const float mnew  = fmaxf(mrun[j], t);
      const float scale = __expf(mrun[j] - mnew);
      mrun[j] = mnew;
      const float e0 = __expf(s0[j] - mnew);
      const float e1 = __expf(s1[j] - mnew);
      p0[j] = e0;
      p1[j] = e1;
      float rs = e0 + e1;
      rs += __shfl_xor(rs, 1, 32);
      rs += __shfl_xor(rs, 2, 32);
      rs += __shfl_xor(rs, 4, 32);
      rs += __shfl_xor(rs, 8, 32);
      lrun[j] = lrun[j] * scale + rs;
      o0[j] *= scale; o1[j] *= scale; o2[j] *= scale; o3[j] *= scale;
    }
    // ---- relayout P: C-layout (lane=col) -> A-layout (lane=row) via LDS ----
#pragma unroll
    for (int j = 0; j < 8; ++j) {
      const int prow = (hs << 3) + j;
      pbuf[wave][prow][r]      = (_Float16)p0[j];
      pbuf[wave][prow][16 + r] = (_Float16)p1[j];
    }
    const v16h pa = frag_from_f16(&pbuf[wave][r][koff]);
    // ---- O += P @ V ----
    o0 = wmma_f16(pa, vb0, o0);
    o1 = wmma_f16(pa, vb1, o1);
    o2 = wmma_f16(pa, vb2, o2);
    o3 = wmma_f16(pa, vb3, o3);
  }

  // ---- normalize and write context [b][q][h*64+d] (f16) ----
#pragma unroll
  for (int j = 0; j < 8; ++j) {
    const int row = q0 + (hs << 3) + j;
    const float inv = 1.0f / lrun[j];
    _Float16* crow = ctx + (size_t)(b * QLEN + row) * DIM + (h << 6) + r;
    crow[0]  = (_Float16)(o0[j] * inv);
    crow[16] = (_Float16)(o1[j] * inv);
    crow[32] = (_Float16)(o2[j] * inv);
    crow[48] = (_Float16)(o3[j] * inv);
  }
}

// ---------------------------------------------------------------------------
// Kernel 3: output projection  out = ctx @ Wo^T + bo  (fp32 out), 16x64/wave.
// ---------------------------------------------------------------------------
__global__ __launch_bounds__(256) void out_proj_kernel(
    const _Float16* __restrict__ ctx, const _Float16* __restrict__ Woh,
    const float* __restrict__ bo, float* __restrict__ out) {
  const int wave = threadIdx.x >> 5;
  const int lane = threadIdx.x & 31;
  const int tile = blockIdx.x * 8 + wave;            // 256 * 16 tiles
  const int m0 = (tile >> 4) << 4;
  const int n0 = (tile & 15) << 6;
  const int r = lane & 15, hs = lane >> 4, koff = hs << 3;

  const _Float16* crow = ctx + (size_t)(m0 + r) * DIM + koff;
  const _Float16* w0   = Woh + (size_t)(n0 + r) * DIM + koff;

  v8f acc0 = {}, acc1 = {}, acc2 = {}, acc3 = {};
  gemm_16x64(crow, w0, acc0, acc1, acc2, acc3);

  auto store_tile = [&](v8f acc, int t) {
    const int n = n0 + t * 16 + r;
    const float bn = bo[n];
#pragma unroll
    for (int j = 0; j < 8; ++j) {
      const int m = m0 + (hs << 3) + j;
      out[(size_t)m * DIM + n] = acc[j] + bn;
    }
  };
  store_tile(acc0, 0);
  store_tile(acc1, 1);
  store_tile(acc2, 2);
  store_tile(acc3, 3);
}

// ---------------------------------------------------------------------------
extern "C" void kernel_launch(void* const* d_in, const int* in_sizes, int n_in,
                              void* d_out, int out_size, void* d_ws, size_t ws_size,
                              hipStream_t stream) {
  const float* X  = (const float*)d_in[0];
  const int* mask = (const int*)d_in[1];
  const float* Wq = (const float*)d_in[2];
  const float* bq = (const float*)d_in[3];
  const float* Wk = (const float*)d_in[4];
  const float* bk = (const float*)d_in[5];
  const float* Wv = (const float*)d_in[6];
  const float* bv = (const float*)d_in[7];
  const float* Wo = (const float*)d_in[8];
  const float* bo = (const float*)d_in[9];
  float* out = (float*)d_out;

  const size_t XN = (size_t)BSZ * QLEN * DIM;        // 4 Mi
  const size_t WN = (size_t)DIM * DIM;               // 1 Mi
  _Float16* Xh    = (_Float16*)d_ws;                 // 8 MB
  _Float16* Wqh   = Xh + XN;                         // 2 MB each
  _Float16* Wkh   = Wqh + WN;
  _Float16* Wvh   = Wkh + WN;
  _Float16* Woh   = Wvh + WN;
  _Float16* qbuf  = Woh + WN;                        // 8 MB each
  _Float16* kbuf  = qbuf + XN;
  _Float16* vtbuf = kbuf + XN;
  _Float16* ctx   = vtbuf + XN;

  // ---- stage 0: precision conversion (once) ----
  cvt_f16_kernel<<<(int)(XN / 8 / 256), 256, 0, stream>>>(X, Xh, (int)(XN / 8));
  cvt_f16_kernel<<<(int)(WN / 8 / 256), 256, 0, stream>>>(Wq, Wqh, (int)(WN / 8));
  cvt_f16_kernel<<<(int)(WN / 8 / 256), 256, 0, stream>>>(Wk, Wkh, (int)(WN / 8));
  cvt_f16_kernel<<<(int)(WN / 8 / 256), 256, 0, stream>>>(Wv, Wvh, (int)(WN / 8));
  cvt_f16_kernel<<<(int)(WN / 8 / 256), 256, 0, stream>>>(Wo, Woh, (int)(WN / 8));

  // ---- stage 1: QKV projection: 3*256*16 tiles / 8 waves = 1536 blocks ----
  qkv_proj_kernel<<<1536, 256, 0, stream>>>(Xh, Wqh, bq, Wkh, bk, Wvh, bv,
                                            qbuf, kbuf, vtbuf);
  // ---- stage 2: attention: 4096 waves / 8 = 512 blocks ----
  attn_kernel<<<512, 256, 0, stream>>>(qbuf, kbuf, vtbuf, mask, ctx);
  // ---- stage 3: output projection: 256*16 tiles / 8 = 512 blocks ----
  out_proj_kernel<<<512, 256, 0, stream>>>(ctx, Woh, bo, out);
}
